// MiniLiquidGPT_53171695125072
// MI455X (gfx1250) — matrix operand order repaired and checked by reading, not compiled, not run on hardware
//
#include <hip/hip_runtime.h>
#include <hip/hip_bf16.h>
#include <math.h>

// ---------------------------------------------------------------------------
// MiniLiquidGPT forward on MI455X (gfx1250, wave32, WMMA).
//   k_init   : zero producer/consumer flags
//   k_embed  : embedding gather + sinusoidal PE -> xemb[T][8][256] f32
//   k_packB  : embed f32 -> bf16, pre-packed WMMA-B tiles (embedB)
//   k_scan   : 4 blocks (one per layer), pipelined over T with global flags.
//              Weights in VGPRs (WMMA-B layout), Hebbian traces in LDS bf16,
//              x/h kept as zero-padded bf16 LDS mirrors (h double-buffered)
//              so every WMMA A/B operand is 2 x ds_load_b128 and each Euler
//              step needs a single barrier. No spills: operands are rebuilt
//              from LDS, only the 128-VGPR weight block stays resident.
//   k_proj   : ys[4096,256]bf16 @ embedB -> out[8][512][50257] f32 (bulk FLOPs)
// ---------------------------------------------------------------------------

constexpr int Vv = 50257, Dd = 256, Bb = 8, Tt = 512;
constexpr int NTILE = (Vv + 15) / 16;          // 3142 n-tiles of 16
constexpr float ETA_B = 0.01f / 8.0f;
constexpr float DECAYC = 0.9f;

// ---- workspace layout (bytes) ----
constexpr size_t OFF_XEMB = 0;
constexpr size_t SZ_XEMB  = (size_t)Tt * Bb * Dd * 4;          // 4 MiB
constexpr size_t OFF_XBUF = OFF_XEMB + SZ_XEMB;
constexpr size_t SZ_XBUF  = (size_t)3 * Tt * Bb * Dd * 4;      // 12 MiB
constexpr size_t OFF_FLAGS = OFF_XBUF + SZ_XBUF;
constexpr size_t SZ_FLAGS  = (size_t)3 * Tt * 4;
constexpr size_t OFF_YS   = (OFF_FLAGS + SZ_FLAGS + 255) & ~(size_t)255;
constexpr size_t SZ_YS    = (size_t)Tt * Bb * Dd * 2;          // 2 MiB bf16
constexpr size_t OFF_EB   = (OFF_YS + SZ_YS + 255) & ~(size_t)255;
constexpr size_t SZ_EB    = (size_t)NTILE * 8 * 32 * 16 * 2;   // 24.5 MiB bf16

// ---- LDS layout for k_scan (bytes) ----
constexpr int L_HEBBI = 0;                 // [256][256] bf16
constexpr int L_HEBBH = 131072;            // [256][256] bf16
constexpr int L_XF    = 262144;            // [8][256]  f32
constexpr int L_HF    = 270336;            // [8][256]  f32
constexpr int L_HPF   = 278528;            // [8][256]  f32 (carry)
constexpr int L_XB    = 286720;            // [16][256] bf16 (rows 8-15 zero)
constexpr int L_HB0   = 294912;            // [16][256] bf16 (rows 8-15 zero)
constexpr int L_HB1   = 303104;            // [16][256] bf16 (rows 8-15 zero)
constexpr int L_LNP   = 311296;            // 4x256 f32: lng, lnb, outg, outb
constexpr int SMEM_SCAN = 315392;

// ---- WMMA types / helpers ----
typedef __attribute__((ext_vector_type(16))) __bf16    v16bf;
typedef __attribute__((ext_vector_type(8)))  float     v8f;
typedef __attribute__((ext_vector_type(4)))  unsigned  v4u;

union BFU { v16bf v; unsigned u32[8]; v4u u4[2]; };

__device__ __forceinline__ unsigned bfbits(float f) {
  unsigned x = __float_as_uint(f);
  return (x + 0x7FFFu + ((x >> 16) & 1u)) >> 16;     // RNE f32 -> bf16
}
__device__ __forceinline__ unsigned pack2(float lo, float hi) {
  return bfbits(lo) | (bfbits(hi) << 16);
}
__device__ __forceinline__ float bf2f(unsigned short h) {
  return __uint_as_float(((unsigned)h) << 16);
}
// B-matrix (16-bit 32x16) K index: lanes 0-15 -> K 0..7,16..23, lanes 16-31 -> 8..15,24..31
__device__ __forceinline__ int jkB(int j, int hf) { return 8 * hf + 2 * j + 8 * (j >> 2); }

__device__ __forceinline__ v8f wmma_bf16(const BFU& a, const BFU& b, v8f c) {
  return __builtin_amdgcn_wmma_f32_16x16x32_bf16(
      false, a.v, false, b.v, (short)0, c, false, false);
}
// A operand from a 16-row bf16 LDS mirror: two b128 loads
__device__ __forceinline__ BFU ldA(const unsigned short* mir, int row, int kt, int hf) {
  const v4u* r = (const v4u*)(mir + row * 256);
  BFU a;
  a.u4[0] = r[4 * kt + hf];
  a.u4[1] = r[4 * kt + 2 + hf];
  return a;
}
// B operand from a bf16 LDS row (hebb row nCol): two b128 loads
__device__ __forceinline__ BFU ldBrow(const unsigned short* rowp, int kt, int hf) {
  const v4u* r = (const v4u*)rowp;
  BFU b;
  b.u4[0] = r[4 * kt + hf];
  b.u4[1] = r[4 * kt + 2 + hf];
  return b;
}

// ---------------------------------------------------------------------------
__global__ void k_init(int* flags) {
  for (int i = threadIdx.x; i < 3 * Tt; i += blockDim.x) flags[i] = 0;
}

// xemb[t][b][d] = embed[ids[b][t]][d] + PE[t][d]
__global__ void k_embed(const int* __restrict__ ids,
                        const float* __restrict__ embed,
                        float* __restrict__ xemb) {
  int i = blockIdx.x * 256 + threadIdx.x;        // < T*8*256
  int d = i & 255, b = (i >> 8) & 7, t = i >> 11;
  int id = ids[b * Tt + t];
  int j = d >> 1;
  float freq = __expf(-(float)(2 * j) * (9.210340371976184f / 256.0f));
  float ang = (float)t * freq;
  float pe = (d & 1) ? __cosf(ang) : __sinf(ang);
  xemb[i] = embed[(size_t)id * Dd + d] + pe;
}

// Pre-pack embed^T into WMMA-B bf16 tiles: embedB[nt][kt][lane][16 bf16]
__global__ void k_packB(const float* __restrict__ embed,
                        unsigned short* __restrict__ embedB) {
  int nt = blockIdx.x;                       // 0..NTILE-1
  int kt = threadIdx.x >> 5, lane = threadIdx.x & 31;
  int ln16 = lane & 15, hf = lane >> 4;
  int v = nt * 16 + ln16;
  BFU w;
#pragma unroll
  for (int j = 0; j < 8; ++j) {
    int k = 32 * kt + jkB(j, hf);
    float f0 = 0.f, f1 = 0.f;
    if (v < Vv) { f0 = embed[(size_t)v * Dd + k]; f1 = embed[(size_t)v * Dd + k + 1]; }
    w.u32[j] = pack2(f0, f1);
  }
  v4u* dst = (v4u*)(embedB + (((size_t)(nt * 8 + kt) * 32 + lane) * 16));
  dst[0] = w.u4[0];
  dst[1] = w.u4[1];
}

// ---------------------------------------------------------------------------
// The sequential scan: blockIdx.x = layer (0..3). 16 waves; wave w owns output
// columns n = 16w..16w+15 and hebb rows 16w..16w+15 (self-consistent).
__global__ __launch_bounds__(512, 1)
void k_scan(const float* __restrict__ xemb,
            const float* __restrict__ Wih, const float* __restrict__ Whh,
            const float* __restrict__ bias, const float* __restrict__ taur,
            const float* __restrict__ lng, const float* __restrict__ lnb,
            const float* __restrict__ outg, const float* __restrict__ outb,
            float* __restrict__ xbuf, int* __restrict__ flags,
            unsigned short* __restrict__ ys) {
  extern __shared__ unsigned char smem[];
  unsigned short* hebbI = (unsigned short*)(smem + L_HEBBI);
  unsigned short* hebbH = (unsigned short*)(smem + L_HEBBH);
  float* xls  = (float*)(smem + L_XF);
  float* hls  = (float*)(smem + L_HF);
  float* hpls = (float*)(smem + L_HPF);
  unsigned short* xbf = (unsigned short*)(smem + L_XB);
  unsigned short* hb0 = (unsigned short*)(smem + L_HB0);
  unsigned short* hb1 = (unsigned short*)(smem + L_HB1);
  float* lnp = (float*)(smem + L_LNP);       // [0..255]=g, [256..511]=b,
                                             // [512..767]=outg, [768..1023]=outb
  const int layer = blockIdx.x;
  const int tid = threadIdx.x, wv = tid >> 5, lane = tid & 31;
  const int ln16 = lane & 15, hf = lane >> 4;
  const int nCol = wv * 16 + ln16;
  const bool plast = (layer >= 2);
  const int steps = plast ? 3 : 1;
  const float dt = 1.0f / (float)steps;
  const int mr = ln16;                 // this lane's A row (rows 8-15 padded 0)

  for (int i = tid; i < 65536; i += 512) { hebbI[i] = 0; hebbH[i] = 0; }
  for (int i = tid; i < 2048; i += 512) hpls[i] = 0.0f;
  for (int i = tid; i < 4096; i += 512) { xbf[i] = 0; hb0[i] = 0; hb1[i] = 0; }
  for (int i = tid; i < 256; i += 512) {
    lnp[i]       = lng[layer * Dd + i];
    lnp[256 + i] = lnb[layer * Dd + i];
    lnp[512 + i] = outg[i];
    lnp[768 + i] = outb[i];
  }

  const float b_n  = bias[layer * Dd + nCol];
  const float taun = logf(1.0f + expf(taur[layer * Dd + nCol])) + 0.1f;
  const float itau = 1.0f / taun;

  // Weights in registers: B-operand layout, K-concat [W_ih | W_hh] (K=512)
  BFU wB[16];
  {
    const float* Wi = Wih + (size_t)layer * 65536 + (size_t)nCol * Dd;
    const float* Wh = Whh + (size_t)layer * 65536 + (size_t)nCol * Dd;
#pragma unroll
    for (int kt = 0; kt < 16; ++kt) {
      const float* W = (kt < 8) ? Wi : Wh;
      const int kb = 32 * (kt & 7);
#pragma unroll
      for (int j = 0; j < 8; ++j) {
        const int k = kb + jkB(j, hf);
        wB[kt].u32[j] = pack2(W[k], W[k + 1]);
      }
    }
  }
  __syncthreads();

  for (int t = 0; t < Tt; ++t) {
    if (layer == 0 && t + 1 < Tt)      // gfx1250 global_prefetch of next slice
      __builtin_prefetch(xemb + (size_t)(t + 1) * 2048 + tid * 4, 0, 3);
    if (layer > 0) {                   // wait for producer (layer-1) at time t
      int* f = flags + (layer - 1) * Tt + t;
      while (__hip_atomic_load(f, __ATOMIC_ACQUIRE, __HIP_MEMORY_SCOPE_AGENT) == 0)
        __builtin_amdgcn_s_sleep(1);
    }
    const float* xsrc = (layer == 0) ? (xemb + (size_t)t * 2048)
                                     : (xbuf + ((size_t)(layer - 1) * Tt + t) * 2048);
    // copy-in: f32 copies + pair-packed bf16 mirrors (rows 0..7)
    for (int i = tid; i < 1024; i += 512) {
      const float2 x2 = ((const float2*)xsrc)[i];
      ((float2*)xls)[i] = x2;
      ((unsigned*)xbf)[i] = pack2(x2.x, x2.y);
      const float2 h2 = ((const float2*)hpls)[i];
      ((float2*)hls)[i] = h2;
      ((unsigned*)hb0)[i] = pack2(h2.x, h2.y);
    }
    __syncthreads();

    if (plast) {
      const unsigned short* rI = hebbI + (size_t)nCol * 256;
      const unsigned short* rH = hebbH + (size_t)nCol * 256;
      for (int s = 0; s < 3; ++s) {
        const unsigned short* hrd = (s & 1) ? hb1 : hb0;
        unsigned short*       hwr = (s & 1) ? hb0 : hb1;
        v8f c0 = {}, c1 = {};
#pragma unroll
        for (int kt = 0; kt < 8; ++kt) {     // x@(Wih+hebbI)^T
          BFU a = ldA(xbf, mr, kt, hf);
          BFU b = ldBrow(rI, kt, hf);
          c0 = wmma_bf16(a, wB[kt], c0);
          c1 = wmma_bf16(a, b, c1);
        }
#pragma unroll
        for (int kt = 0; kt < 8; ++kt) {     // h@(Whh+hebbH)^T
          BFU a = ldA(hrd, mr, kt, hf);
          BFU b = ldBrow(rH, kt, hf);
          c0 = wmma_bf16(a, wB[8 + kt], c0);
          c1 = wmma_bf16(a, b, c1);
        }
        const v8f c = c0 + c1;
        if (hf == 0) {                       // single-owner RMW of h
#pragma unroll
          for (int r = 0; r < 8; ++r) {
            const float hp = hls[r * 256 + nCol];
            const float pre = c[r] + b_n;
            const float hn = hp + dt * (tanhf(pre) - hp) * itau;
            hls[r * 256 + nCol] = hn;
            hwr[r * 256 + nCol] = (unsigned short)bfbits(hn);
          }
        }
        __syncthreads();                     // one barrier per Euler step
      }
    } else {
      v8f c0 = {}, c1 = {};
#pragma unroll
      for (int kt = 0; kt < 8; ++kt) {
        BFU a = ldA(xbf, mr, kt, hf);
        if (kt & 1) c1 = wmma_bf16(a, wB[kt], c1);
        else        c0 = wmma_bf16(a, wB[kt], c0);
      }
#pragma unroll
      for (int kt = 0; kt < 8; ++kt) {
        BFU a = ldA(hb0, mr, kt, hf);
        if (kt & 1) c1 = wmma_bf16(a, wB[8 + kt], c1);
        else        c0 = wmma_bf16(a, wB[8 + kt], c0);
      }
      const v8f c = c0 + c1;
      if (hf == 0) {
#pragma unroll
        for (int r = 0; r < 8; ++r) {
          const float hp = hls[r * 256 + nCol];
          const float pre = c[r] + b_n;
          const float hn = hp + (tanhf(pre) - hp) * itau;   // dt = 1
          hls[r * 256 + nCol] = hn;
        }
      }
      __syncthreads();
    }

    if (plast) {  // hebb = 0.9*hebb + (eta/B) h^T {x | h_prev}, K=8 padded
      BFU aH;     // A = (eta/B)*h^T, wave w owns p-tile w, p = nCol
#pragma unroll
      for (int j = 0; j < 8; ++j) {
        if (j < 4) {
          const unsigned pv = pack2(ETA_B * hls[(2 * j) * 256 + nCol],
                                    ETA_B * hls[(2 * j + 1) * 256 + nCol]);
          aH.u32[j] = (hf == 0) ? pv : 0u;
        } else aH.u32[j] = 0u;
      }
      for (int qt = 0; qt < 16; ++qt) {
        const int q = 16 * qt + ln16;
        BFU bX, bH;
#pragma unroll
        for (int j = 0; j < 8; ++j) {
          if (j < 4) {
            const unsigned vx = pack2(xls[(2 * j) * 256 + q],  xls[(2 * j + 1) * 256 + q]);
            const unsigned vh = pack2(hpls[(2 * j) * 256 + q], hpls[(2 * j + 1) * 256 + q]);
            bX.u32[j] = (hf == 0) ? vx : 0u;
            bH.u32[j] = (hf == 0) ? vh : 0u;
          } else { bX.u32[j] = 0u; bH.u32[j] = 0u; }
        }
        v8f cI, cH;
#pragma unroll
        for (int r = 0; r < 8; ++r) {
          const int p = 16 * wv + r + 8 * hf;
          cI[r] = DECAYC * bf2f(hebbI[p * 256 + q]);
          cH[r] = DECAYC * bf2f(hebbH[p * 256 + q]);
        }
        cI = wmma_bf16(aH, bX, cI);
        cH = wmma_bf16(aH, bH, cH);
#pragma unroll
        for (int r = 0; r < 8; ++r) {
          const int p = 16 * wv + r + 8 * hf;
          hebbI[p * 256 + q] = (unsigned short)bfbits(cI[r]);
          hebbH[p * 256 + q] = (unsigned short)bfbits(cH[r]);
        }
      }
    }
    __syncthreads();   // protect hpls: hebb readers done before LN writers

    // residual + LayerNorm; waves 0..7 each own one batch row
    if (wv < 8) {
      const int m = wv;
      float v[8], sum = 0.f, sq = 0.f;
#pragma unroll
      for (int j = 0; j < 8; ++j) {
        const int n = lane + 32 * j;
        v[j] = hls[m * 256 + n] + xls[m * 256 + n];
        sum += v[j]; sq += v[j] * v[j];
      }
#pragma unroll
      for (int o = 16; o > 0; o >>= 1) { sum += __shfl_xor(sum, o, 32); sq += __shfl_xor(sq, o, 32); }
      const float mean = sum * (1.f / 256.f);
      const float rstd = rsqrtf(sq * (1.f / 256.f) - mean * mean + 1e-5f);
      if (layer < 3) {
        float* dst = xbuf + ((size_t)layer * Tt + t) * 2048 + m * 256;
#pragma unroll
        for (int j = 0; j < 8; ++j) {
          const int n = lane + 32 * j;
          const float y = (v[j] - mean) * rstd * lnp[n] + lnp[256 + n];
          dst[n] = y;
          hpls[m * 256 + n] = y;
        }
      } else {            // layer 3: carry LN, then output LN -> ys (bf16)
        float u[8], s2 = 0.f, q2 = 0.f;
#pragma unroll
        for (int j = 0; j < 8; ++j) {
          const int n = lane + 32 * j;
          u[j] = (v[j] - mean) * rstd * lnp[n] + lnp[256 + n];
          hpls[m * 256 + n] = u[j];
          s2 += u[j]; q2 += u[j] * u[j];
        }
#pragma unroll
        for (int o = 16; o > 0; o >>= 1) { s2 += __shfl_xor(s2, o, 32); q2 += __shfl_xor(q2, o, 32); }
        const float mean2 = s2 * (1.f / 256.f);
        const float rstd2 = rsqrtf(q2 * (1.f / 256.f) - mean2 * mean2 + 1e-5f);
#pragma unroll
        for (int j = 0; j < 8; ++j) {
          const int n = lane + 32 * j;
          const float y = (u[j] - mean2) * rstd2 * lnp[512 + n] + lnp[768 + n];
          ys[((size_t)t * 8 + m) * 256 + n] = (unsigned short)bfbits(y);
        }
      }
    }
    if (layer < 3) {
      __threadfence();
      __syncthreads();
      if (tid == 0)
        __hip_atomic_store(flags + layer * Tt + t, 1, __ATOMIC_RELEASE,
                           __HIP_MEMORY_SCOPE_AGENT);
    } else {
      __syncthreads();
    }
  }
}

// ---------------------------------------------------------------------------
// out[b][t][v] = sum_k ys[t*8+b][k] * embed[v][k]; M=4096, N=50257, K=256
// Each wave computes one 16-row A slab against two n-tiles (A reused).
__global__ __launch_bounds__(256)
void k_proj(const unsigned short* __restrict__ ys,
            const unsigned short* __restrict__ embedB,
            float* __restrict__ out) {
  const int wv = threadIdx.x >> 5, lane = threadIdx.x & 31;
  const int ln16 = lane & 15, hf = lane >> 4;
  const int mt = blockIdx.x;                  // 0..255
  const int nt0 = blockIdx.y * 16 + wv;
  const int nt1 = nt0 + 8;
  if (nt0 >= NTILE) return;
  const bool p1 = (nt1 < NTILE);

  const int m = mt * 16 + ln16;
  const v4u* yrow = (const v4u*)(ys + (size_t)m * 256);
  BFU a[8];
#pragma unroll
  for (int kt = 0; kt < 8; ++kt) {
    a[kt].u4[0] = yrow[4 * kt + hf];
    a[kt].u4[1] = yrow[4 * kt + 2 + hf];
  }
  v8f c0 = {}, c1 = {}, c2 = {}, c3 = {};
  const v4u* bp0 = (const v4u*)(embedB + (size_t)nt0 * 8 * 32 * 16);
#pragma unroll
  for (int kt = 0; kt < 8; ++kt) {
    BFU b;
    const v4u* src = bp0 + ((size_t)kt * 32 + lane) * 2;
    b.u4[0] = src[0];
    b.u4[1] = src[1];
    if (kt & 1) c1 = wmma_bf16(a[kt], b, c1);
    else        c0 = wmma_bf16(a[kt], b, c0);
  }
  if (p1) {
    const v4u* bp1 = (const v4u*)(embedB + (size_t)nt1 * 8 * 32 * 16);
#pragma unroll
    for (int kt = 0; kt < 8; ++kt) {
      BFU b;
      const v4u* src = bp1 + ((size_t)kt * 32 + lane) * 2;
      b.u4[0] = src[0];
      b.u4[1] = src[1];
      if (kt & 1) c3 = wmma_bf16(a[kt], b, c3);
      else        c2 = wmma_bf16(a[kt], b, c2);
    }
  }
  const v8f cA = c0 + c1;
  const int v0 = nt0 * 16 + ln16;
  if (v0 < Vv) {
#pragma unroll
    for (int r = 0; r < 8; ++r) {
      const int mm = mt * 16 + r + 8 * hf;
      const int tt = mm >> 3, bb = mm & 7;
      __builtin_nontemporal_store(cA[r], &out[((size_t)bb * Tt + tt) * Vv + v0]);
    }
  }
  if (p1) {
    const v8f cB = c2 + c3;
    const int v1 = nt1 * 16 + ln16;
    if (v1 < Vv) {
#pragma unroll
      for (int r = 0; r < 8; ++r) {
        const int mm = mt * 16 + r + 8 * hf;
        const int tt = mm >> 3, bb = mm & 7;
        __builtin_nontemporal_store(cB[r], &out[((size_t)bb * Tt + tt) * Vv + v1]);
      }
    }
  }
}

// ---------------------------------------------------------------------------
extern "C" void kernel_launch(void* const* d_in, const int* in_sizes, int n_in,
                              void* d_out, int out_size, void* d_ws, size_t ws_size,
                              hipStream_t stream) {
  (void)in_sizes; (void)n_in; (void)out_size; (void)ws_size;
  const int*   ids   = (const int*)  d_in[0];
  const float* embed = (const float*)d_in[1];
  const float* Wih   = (const float*)d_in[2];
  const float* Whh   = (const float*)d_in[3];
  const float* bias  = (const float*)d_in[4];
  const float* taur  = (const float*)d_in[5];
  const float* lng   = (const float*)d_in[6];
  const float* lnb   = (const float*)d_in[7];
  const float* outg  = (const float*)d_in[8];
  const float* outb  = (const float*)d_in[9];
  float* out = (float*)d_out;

  char* ws = (char*)d_ws;
  float*          xemb   = (float*)(ws + OFF_XEMB);
  float*          xbuf   = (float*)(ws + OFF_XBUF);
  int*            flags  = (int*)  (ws + OFF_FLAGS);
  unsigned short* ys     = (unsigned short*)(ws + OFF_YS);
  unsigned short* embedB = (unsigned short*)(ws + OFF_EB);

  (void)hipFuncSetAttribute((const void*)k_scan,
                            hipFuncAttributeMaxDynamicSharedMemorySize, SMEM_SCAN);

  k_init<<<1, 512, 0, stream>>>(flags);
  k_embed<<<(Tt * Bb * Dd) / 256, 256, 0, stream>>>(ids, embed, xemb);
  k_packB<<<NTILE, 256, 0, stream>>>(embed, embedB);
  k_scan<<<4, 512, SMEM_SCAN, stream>>>(xemb, Wih, Whh, bias, taur,
                                        lng, lnb, outg, outb, xbuf, flags, ys);
  dim3 pg(256, (NTILE + 15) / 16);
  k_proj<<<pg, 256, 0, stream>>>(ys, embedB, out);
}
